// ChittaEncoder_71098888618432
// MI455X (gfx1250) — matrix operand order repaired and checked by reading, not compiled, not run on hardware
//
#include <hip/hip_runtime.h>

typedef __bf16 bf16_t;
typedef __bf16 v8bf  __attribute__((ext_vector_type(8)));
typedef __bf16 v16bf __attribute__((ext_vector_type(16)));
typedef float  v8f   __attribute__((ext_vector_type(8)));
typedef int    v4i   __attribute__((ext_vector_type(4)));

#define B_DIM 2048
#define D_DIM 2048
#define N_DIM 32768
#define K_SEL 32
#define LSTR  40                            // LDS tile row stride (bf16), padded vs bank conflicts
#define OSTR  136                           // LDS output-staging row stride (bf16)
#define SCORE_SCALE 0.022097086912079608f   // 1/sqrt(2048)

// ---------------------------------------------------------------- CDNA5 async-to-LDS path
#if __has_builtin(__builtin_amdgcn_global_load_async_to_lds_b128) && \
    __has_builtin(__builtin_amdgcn_s_wait_asynccnt)
#define HAS_ASYNC_LDS 1
#warning CDNA5_PROBE: global_load_async_to_lds_b128 builtin AVAILABLE - using async pipeline
#else
#define HAS_ASYNC_LDS 0
#warning CDNA5_PROBE: global_load_async_to_lds_b128 builtin MISSING - using sync staging fallback
#endif

typedef __attribute__((address_space(1))) v4i* as1_v4i;   // global int4*
typedef __attribute__((address_space(3))) v4i* as3_v4i;   // LDS int4*

__device__ inline void async_wait0() {
#if HAS_ASYNC_LDS
    __builtin_amdgcn_s_wait_asynccnt(0);
#endif
}

// copy 16 bf16 (32B) global -> LDS for this thread
__device__ inline void stage16(const bf16_t* __restrict__ g, bf16_t* l) {
#if HAS_ASYNC_LDS
    __builtin_amdgcn_global_load_async_to_lds_b128(
        (as1_v4i)(uintptr_t)g, (as3_v4i)(uintptr_t)l, 0, 0);
    __builtin_amdgcn_global_load_async_to_lds_b128(
        (as1_v4i)(uintptr_t)(g + 8), (as3_v4i)(uintptr_t)(l + 8), 0, 0);
#else
    const uint4* gp = (const uint4*)g;
    uint4* lp = (uint4*)l;
    lp[0] = gp[0]; lp[1] = gp[1];
#endif
}

// ---------------------------------------------------------------- WMMA helpers
__device__ inline v8f wmma_bf16(v16bf a, v16bf b, v8f c) {
    return __builtin_amdgcn_wmma_f32_16x16x32_bf16(false, a, false, b, (short)0, c, false, false);
}

// A fragment: lane l -> row (l&15); elems 0..7 = K[kh*8..], 8..15 = K[16+kh*8..], kh=l>>4
__device__ inline v16bf frag_a(const bf16_t* base, int lane) {
    const bf16_t* p = base + (lane & 15) * LSTR + (lane >> 4) * 8;
    v8bf lo = *(const v8bf*)p;
    v8bf hi = *(const v8bf*)(p + 16);
    v16bf f;
#pragma unroll
    for (int i = 0; i < 8; ++i) { f[i] = lo[i]; f[i + 8] = hi[i]; }
    return f;
}

// B fragment: lane l -> col (l&15); elems 0..7 = K[kh*8..], 8..15 = K[16+kh*8..], kh=l>>4
__device__ inline v16bf frag_b(const bf16_t* base, int lane) {
    const bf16_t* p = base + (lane & 15) * LSTR + (lane >> 4) * 8;
    v8bf lo = *(const v8bf*)p;
    v8bf hi = *(const v8bf*)(p + 16);
    v16bf f;
#pragma unroll
    for (int i = 0; i < 8; ++i) { f[i] = lo[i]; f[i + 8] = hi[i]; }
    return f;
}

// ---------------------------------------------------------------- kernel 1: karma bias
__global__ __launch_bounds__(1024) void karma_bias_kernel(const float* __restrict__ karma,
                                                          float* __restrict__ bias) {
    __shared__ float red[1024];
    const int t = threadIdx.x;
    float mx = -1e30f;
    for (int i = t; i < N_DIM; i += 1024) mx = fmaxf(mx, karma[i]);
    red[t] = mx; __syncthreads();
    for (int s = 512; s > 0; s >>= 1) { if (t < s) red[t] = fmaxf(red[t], red[t + s]); __syncthreads(); }
    const float gmax = red[0]; __syncthreads();
    float sm = 0.f;
    for (int i = t; i < N_DIM; i += 1024) sm += expf(karma[i] - gmax);
    red[t] = sm; __syncthreads();
    for (int s = 512; s > 0; s >>= 1) { if (t < s) red[t] += red[t + s]; __syncthreads(); }
    const float logZ = gmax + logf(red[0]);
    for (int i = t; i < N_DIM; i += 1024) bias[i] = fmaxf(karma[i] - logZ, -10.0f);
}

// ---------------------------------------------------------------- kernel 2: f32 -> bf16 cast
__global__ __launch_bounds__(256) void cast_f32_to_bf16(const float* __restrict__ src,
                                                        bf16_t* __restrict__ dst, int n4) {
    int i = blockIdx.x * 256 + threadIdx.x;
    if (i < n4) {
        float4 v = ((const float4*)src)[i];
        bf16_t o[4] = { (bf16_t)v.x, (bf16_t)v.y, (bf16_t)v.z, (bf16_t)v.w };
        *(uint2*)(dst + (size_t)i * 4) = *(uint2*)o;
    }
}

// ---------------------------------------------------------------- kernel 3: q = x @ W^T (bf16 WMMA)
// grid (D/128, B/128), 256 threads = 8 waves; wave -> 32x64 output; double-buffered LDS pipeline.
__global__ __launch_bounds__(256) void qproj_wmma(const bf16_t* __restrict__ X,
                                                  const bf16_t* __restrict__ W,
                                                  bf16_t* __restrict__ Q) {
    __shared__ bf16_t lA[2][128 * LSTR];
    __shared__ bf16_t lB[2][128 * LSTR];
    __shared__ bf16_t lOut[128 * OSTR];
    const int t = threadIdx.x;
    const int lane = t & 31, wave = t >> 5;
    const int m0 = blockIdx.y * 128, n0 = blockIdx.x * 128;
    const int wm = (wave & 3) * 32, wn = (wave >> 2) * 64;
    const int lr = t >> 1, lh = (t & 1) * 16;
    const int stoff = lr * LSTR + lh;
    const bf16_t* gA = X + (size_t)(m0 + lr) * D_DIM + lh;
    const bf16_t* gB = W + (size_t)(n0 + lr) * D_DIM + lh;
    v8f acc[2][4] = {};

    stage16(gA, &lA[0][stoff]);
    stage16(gB, &lB[0][stoff]);
    async_wait0();
    __syncthreads();

    int buf = 0;
    for (int k0 = 0; k0 < D_DIM; k0 += 32) {
        if (k0 + 32 < D_DIM) {                    // prefetch next tile into other buffer
            stage16(gA + k0 + 32, &lA[buf ^ 1][stoff]);
            stage16(gB + k0 + 32, &lB[buf ^ 1][stoff]);
        }
        v16bf afr[2], bfr[4];
#pragma unroll
        for (int mi = 0; mi < 2; ++mi) afr[mi] = frag_a(&lA[buf][(wm + mi * 16) * LSTR], lane);
#pragma unroll
        for (int ni = 0; ni < 4; ++ni) bfr[ni] = frag_b(&lB[buf][(wn + ni * 16) * LSTR], lane);
#pragma unroll
        for (int mi = 0; mi < 2; ++mi)
#pragma unroll
            for (int ni = 0; ni < 4; ++ni)
                acc[mi][ni] = wmma_bf16(afr[mi], bfr[ni], acc[mi][ni]);
        async_wait0();
        __syncthreads();
        buf ^= 1;
    }

    // epilogue: stage output tile in LDS, then coalesced b128 stores
    const int cr = (lane >> 4) * 8, cc = lane & 15;
#pragma unroll
    for (int mi = 0; mi < 2; ++mi)
#pragma unroll
        for (int ni = 0; ni < 4; ++ni)
#pragma unroll
            for (int v = 0; v < 8; ++v)
                lOut[(wm + mi * 16 + cr + v) * OSTR + wn + ni * 16 + cc] = (bf16_t)acc[mi][ni][v];
    __syncthreads();
    for (int i = t; i < 128 * 8; i += 256) {
        const int row = i >> 3, j = i & 7;
        *(uint4*)(Q + (size_t)(m0 + row) * D_DIM + n0 + j * 16) =
            *(const uint4*)(lOut + row * OSTR + j * 16);
    }
}

// ---------------------------------------------------------------- kernel 4: scores GEMM + streaming top-32
// grid (B/128 row-tiles, N/2048 chunks); per 128-col sub-tile: pipelined WMMA GEMM over K=2048,
// scale+bias into LDS, per-row top-32 update; partials to workspace.
__global__ __launch_bounds__(256) void scores_topk(const bf16_t* __restrict__ Q,
                                                   const bf16_t* __restrict__ S,
                                                   const float* __restrict__ bias,
                                                   float* __restrict__ pscore,
                                                   int* __restrict__ pidx) {
    __shared__ bf16_t lA[2][128 * LSTR];
    __shared__ bf16_t lB[2][128 * LSTR];
    __shared__ float  sc[128][132];
    __shared__ float  topv[128][32];
    __shared__ int    topi[128][32];
    __shared__ float  bbuf[128];
    const int t = threadIdx.x;
    const int lane = t & 31, wave = t >> 5;
    const int m0 = blockIdx.x * 128;
    const int chunk = blockIdx.y;
    const int wm = (wave & 3) * 32, wn = (wave >> 2) * 64;
    const int lr = t >> 1, lh = (t & 1) * 16;
    const int stoff = lr * LSTR + lh;
    const int cr = (lane >> 4) * 8, cc = lane & 15;
    const bf16_t* gA = Q + (size_t)(m0 + lr) * D_DIM + lh;

    for (int i = t; i < 128 * 32; i += 256) { (&topv[0][0])[i] = -1e30f; (&topi[0][0])[i] = 0; }
    float minv = -1e30f; int minp = 0;   // valid for t < 128 (row owner)

    for (int sub = 0; sub < 16; ++sub) {
        const int n0 = chunk * 2048 + sub * 128;
        if (t < 128) bbuf[t] = bias[n0 + t];
        const bf16_t* gB = S + (size_t)(n0 + lr) * D_DIM + lh;
        v8f acc[2][4] = {};

        stage16(gA, &lA[0][stoff]);
        stage16(gB, &lB[0][stoff]);
        async_wait0();
        __syncthreads();

        int buf = 0;
        for (int k0 = 0; k0 < D_DIM; k0 += 32) {
            if (k0 + 32 < D_DIM) {
                stage16(gA + k0 + 32, &lA[buf ^ 1][stoff]);
                stage16(gB + k0 + 32, &lB[buf ^ 1][stoff]);
            }
            v16bf afr[2], bfr[4];
#pragma unroll
            for (int mi = 0; mi < 2; ++mi) afr[mi] = frag_a(&lA[buf][(wm + mi * 16) * LSTR], lane);
#pragma unroll
            for (int ni = 0; ni < 4; ++ni) bfr[ni] = frag_b(&lB[buf][(wn + ni * 16) * LSTR], lane);
#pragma unroll
            for (int mi = 0; mi < 2; ++mi)
#pragma unroll
                for (int ni = 0; ni < 4; ++ni)
                    acc[mi][ni] = wmma_bf16(afr[mi], bfr[ni], acc[mi][ni]);
            async_wait0();
            __syncthreads();
            buf ^= 1;
        }

#pragma unroll
        for (int mi = 0; mi < 2; ++mi)
#pragma unroll
            for (int ni = 0; ni < 4; ++ni)
#pragma unroll
                for (int v = 0; v < 8; ++v)
                    sc[wm + mi * 16 + cr + v][wn + ni * 16 + cc] =
                        acc[mi][ni][v] * SCORE_SCALE + bbuf[wn + ni * 16 + cc];
        __syncthreads();
        if (t < 128) {
            for (int c = 0; c < 128; ++c) {
                const float s = sc[t][c];
                if (s > minv) {
                    topv[t][minp] = s; topi[t][minp] = n0 + c;
                    float mv = topv[t][0]; int mp = 0;
#pragma unroll 4
                    for (int j = 1; j < 32; ++j) { const float tv = topv[t][j]; if (tv < mv) { mv = tv; mp = j; } }
                    minv = mv; minp = mp;
                }
            }
        }
        __syncthreads();
    }
    if (t < 128) {
#pragma unroll 4
        for (int j = 0; j < 32; ++j) {
            pscore[((size_t)(m0 + t) * 16 + chunk) * 32 + j] = topv[t][j];
            pidx  [((size_t)(m0 + t) * 16 + chunk) * 32 + j] = topi[t][j];
        }
    }
}

// ---------------------------------------------------------------- kernel 5: merge 512->32, softmax, field
__global__ __launch_bounds__(256) void merge_topk(const float* __restrict__ pscore,
                                                  const int* __restrict__ pidx,
                                                  const float* __restrict__ seeds,
                                                  float* __restrict__ field,
                                                  float* __restrict__ attn_out) {
    __shared__ float cv[512];
    __shared__ int   ci[512];
    __shared__ float rv[256];
    __shared__ int   ri[256];
    __shared__ float selv[32];
    __shared__ int   seli[32];
    __shared__ float attn[32];
    const int b = blockIdx.x, t = threadIdx.x;
    for (int i = t; i < 512; i += 256) {
        cv[i] = pscore[(size_t)b * 512 + i];
        ci[i] = pidx[(size_t)b * 512 + i];
    }
    __syncthreads();
    for (int k = 0; k < 32; ++k) {           // descending selection; ties -> lower seed idx
        const float v0 = cv[t], v1 = cv[t + 256];
        const bool tk = (v1 > v0) || (v1 == v0 && ci[t + 256] < ci[t]);
        rv[t] = tk ? v1 : v0; ri[t] = tk ? (t + 256) : t;
        __syncthreads();
        for (int s = 128; s > 0; s >>= 1) {
            if (t < s) {
                const float a = rv[t], c = rv[t + s];
                const int ia = ri[t], ic = ri[t + s];
                if ((c > a) || (c == a && ci[ic] < ci[ia])) { rv[t] = c; ri[t] = ic; }
            }
            __syncthreads();
        }
        if (t == 0) { selv[k] = rv[0]; seli[k] = ci[ri[0]]; cv[ri[0]] = -1e38f; }
        __syncthreads();
    }
    if (t < 32) attn[t] = expf(selv[t] - selv[0]);
    __syncthreads();
    if (t == 0) {
        float s = 0.f;
        for (int k = 0; k < 32; ++k) s += attn[k];
        const float inv = 1.f / s;
        for (int k = 0; k < 32; ++k) attn[k] *= inv;
    }
    __syncthreads();
    if (t < 32) attn_out[(size_t)b * 32 + t] = attn[t];
    for (int c = t; c < D_DIM; c += 256) {
        float acc = 0.f;
#pragma unroll 8
        for (int k = 0; k < 32; ++k) acc += attn[k] * seeds[(size_t)seli[k] * D_DIM + c];
        field[(size_t)b * D_DIM + c] = acc;
    }
}

// ---------------------------------------------------------------- launch
extern "C" void kernel_launch(void* const* d_in, const int* in_sizes, int n_in,
                              void* d_out, int out_size, void* d_ws, size_t ws_size,
                              hipStream_t stream) {
    const float* x     = (const float*)d_in[0];
    const float* Wq    = (const float*)d_in[1];
    const float* seeds = (const float*)d_in[2];
    const float* karma = (const float*)d_in[3];

    float* out   = (float*)d_out;
    float* field = out;                                  // [B, D]
    float* attn  = out + (size_t)B_DIM * D_DIM;          // [B, K]

    char* ws = (char*)d_ws;
    float*  kb  = (float*)(ws);                          // [0, 128KB)  karma bias
    bf16_t* xbf = (bf16_t*)(ws + (1ull  << 20));         // [1MB, 9MB)
    bf16_t* wbf = (bf16_t*)(ws + (9ull  << 20));         // [9MB, 17MB)
    bf16_t* qbf = (bf16_t*)(ws + (17ull << 20));         // [17MB, 25MB)
    float*  psc = (float*) (ws + (25ull << 20));         // [25MB, 29MB) partial scores
    int*    pid = (int*)   (ws + (29ull << 20));         // [29MB, 33MB) partial idx
    bf16_t* sbf = (bf16_t*)(ws + (33ull << 20));         // [33MB, 161MB) seeds bf16

    karma_bias_kernel<<<1, 1024, 0, stream>>>(karma, kb);

    const int nx4 = B_DIM * D_DIM / 4;
    const int ns4 = N_DIM * D_DIM / 4;
    cast_f32_to_bf16<<<(nx4 + 255) / 256, 256, 0, stream>>>(x, xbf, nx4);
    cast_f32_to_bf16<<<(nx4 + 255) / 256, 256, 0, stream>>>(Wq, wbf, nx4);
    cast_f32_to_bf16<<<(ns4 + 255) / 256, 256, 0, stream>>>(seeds, sbf, ns4);

    qproj_wmma<<<dim3(D_DIM / 128, B_DIM / 128), 256, 0, stream>>>(xbf, wbf, qbf);

    scores_topk<<<dim3(B_DIM / 128, N_DIM / 2048), 256, 0, stream>>>(qbf, sbf, kb, psc, pid);

    merge_topk<<<B_DIM, 256, 0, stream>>>(psc, pid, seeds, field, attn);
}
